// _ClassicalAgg_14456859918885
// MI455X (gfx1250) — matrix operand order repaired and checked by reading, not compiled, not run on hardware
//
#include <hip/hip_runtime.h>
#include <hip/hip_bf16.h>
#include <math.h>

// Problem constants (match reference)
#define IN_DIM 512
#define QK 64
#define DV 64
#define BH 16
#define SEQ 2048
#define PSTR 136  // LDS row stride in halves (128 + 8 pad, keeps 16B alignment)

typedef _Float16 half_t;
typedef __attribute__((ext_vector_type(16))) _Float16 v16h;
typedef __attribute__((ext_vector_type(8)))  _Float16 v8h;
typedef __attribute__((ext_vector_type(8)))  float    v8f;
typedef unsigned int u32x4 __attribute__((ext_vector_type(4)));
typedef int          i32x8 __attribute__((ext_vector_type(8)));
typedef int          i32x4 __attribute__((ext_vector_type(4)));

#define WMMA_F16(A, B, C) \
  __builtin_amdgcn_wmma_f32_16x16x32_f16(false, (A), false, (B), (short)0, (C), false, false)

// A-frag (16x32 f16): lane m (0-15) holds row m halves {c..c+7, c+16..c+23};
// lane m+16 holds row m halves {c+8..c+15, c+24..c+31}.
static __device__ __forceinline__ v16h load_a_h(const half_t* row, int c, int hi) {
  v16h a;
  ((v8h*)&a)[0] = *(const v8h*)(row + c + hi * 8);
  ((v8h*)&a)[1] = *(const v8h*)(row + c + 16 + hi * 8);
  return a;
}

// B-frag (32x16 f16): lane n (0-15) holds column n, K = {c..c+7, c+16..c+23}; lane n+16
// holds column n, K = {c+8..c+15, c+24..c+31}. Caller passes the column's contiguous data.
static __device__ __forceinline__ v16h load_b_h(const half_t* col, int c, int hi) {
  v16h b;
  ((v8h*)&b)[0] = *(const v8h*)(col + c + hi * 8);
  ((v8h*)&b)[1] = *(const v8h*)(col + c + 16 + hi * 8);
  return b;
}

// A-frag built on the fly from an fp32 row (projection inputs).
static __device__ __forceinline__ v16h load_a_f32(const float* row, int c, int hi) {
  v16h a;
  const float* p0 = row + c + hi * 8;
  const float* p1 = row + c + 16 + hi * 8;
#pragma unroll
  for (int i = 0; i < 8; ++i) {
    a[i]     = (half_t)p0[i];
    a[8 + i] = (half_t)p1[i];
  }
  return a;
}

// Reductions across the 16-lane half-wave groups (xor masks < 16 stay in group).
static __device__ __forceinline__ float redadd16(float x) {
#pragma unroll
  for (int m = 1; m < 16; m <<= 1) x += __shfl_xor(x, m, 32);
  return x;
}
static __device__ __forceinline__ float redmax16(float x) {
#pragma unroll
  for (int m = 1; m < 16; m <<= 1) x = fmaxf(x, __shfl_xor(x, m, 32));
  return x;
}

// ---------------------------------------------------------------------------
// Kernel 0: transpose Wq/Wk to (64 x 512) f16, convert xh_value (b,d,s) -> f16.
// ---------------------------------------------------------------------------
__global__ void _ca_prep_kernel(const float* __restrict__ Wq, const float* __restrict__ Wk,
                                const float* __restrict__ xv,
                                half_t* __restrict__ wqt, half_t* __restrict__ wkt,
                                half_t* __restrict__ vt, int nv) {
  int i = blockIdx.x * 256 + threadIdx.x;
  if (i < nv) vt[i] = (half_t)xv[i];
  if (i < QK * IN_DIM) {
    int n = i >> 9, k = i & 511;         // wt[n*512 + k] = W[k*64 + n]
    wqt[i] = (half_t)Wq[k * QK + n];
    wkt[i] = (half_t)Wk[k * QK + n];
  }
}

// ---------------------------------------------------------------------------
// Kernel 1: Y = LN(X @ W + bias) in f16, plus per-row sum(Y^2).
// Block = 256 threads = 8 waves, each wave does a 16x64 output tile via WMMA.
// ---------------------------------------------------------------------------
__global__ __launch_bounds__(256) void _ca_proj_ln_kernel(
    const float* __restrict__ X,      // 32768 x 512
    const half_t* __restrict__ Wt,    // 64 x 512 (pre-transposed)
    const float* __restrict__ bias,   // 64
    const float* __restrict__ g, const float* __restrict__ bb,  // LN gamma/beta
    half_t* __restrict__ Y,           // 32768 x 64 f16
    float* __restrict__ nrm) {        // 32768
  int w = threadIdx.x >> 5, l = threadIdx.x & 31;
  int lo = l & 15, hi = l >> 4;
  int rowbase = blockIdx.x * 128 + w * 16;
  const float* xrow = X + (size_t)(rowbase + lo) * IN_DIM;

  v8f acc0 = {}, acc1 = {}, acc2 = {}, acc3 = {};
  for (int c = 0; c < IN_DIM; c += 32) {
    v16h a = load_a_f32(xrow, c, hi);
    acc0 = WMMA_F16(a, load_b_h(Wt + (size_t)(0  + lo) * IN_DIM, c, hi), acc0);
    acc1 = WMMA_F16(a, load_b_h(Wt + (size_t)(16 + lo) * IN_DIM, c, hi), acc1);
    acc2 = WMMA_F16(a, load_b_h(Wt + (size_t)(32 + lo) * IN_DIM, c, hi), acc2);
    acc3 = WMMA_F16(a, load_b_h(Wt + (size_t)(48 + lo) * IN_DIM, c, hi), acc3);
  }
  float bi0 = bias[lo], bi1 = bias[16 + lo], bi2 = bias[32 + lo], bi3 = bias[48 + lo];
  float g0 = g[lo],  g1 = g[16 + lo],  g2 = g[32 + lo],  g3 = g[48 + lo];
  float b0 = bb[lo], b1 = bb[16 + lo], b2 = bb[32 + lo], b3 = bb[48 + lo];

#pragma unroll
  for (int r = 0; r < 8; ++r) {
    // C-layout: element r <-> rows (r | r+8) for (lo | hi) half-wave; col = lo + 16*ng.
    float x0 = acc0[r] + bi0, x1 = acc1[r] + bi1, x2 = acc2[r] + bi2, x3 = acc3[r] + bi3;
    float s  = redadd16(x0 + x1 + x2 + x3);
    float ss = redadd16(x0 * x0 + x1 * x1 + x2 * x2 + x3 * x3);
    float mean = s * (1.0f / 64.0f);
    float var  = ss * (1.0f / 64.0f) - mean * mean;
    float rstd = rsqrtf(var + 1e-5f);
    float y0 = (x0 - mean) * rstd * g0 + b0;
    float y1 = (x1 - mean) * rstd * g1 + b1;
    float y2 = (x2 - mean) * rstd * g2 + b2;
    float y3 = (x3 - mean) * rstd * g3 + b3;
    int row = rowbase + r + hi * 8;
    half_t* yr = Y + (size_t)row * QK;
    yr[lo] = (half_t)y0; yr[16 + lo] = (half_t)y1;
    yr[32 + lo] = (half_t)y2; yr[48 + lo] = (half_t)y3;
    float qn = redadd16(y0 * y0 + y1 * y1 + y2 * y2 + y3 * y3);
    if (lo == 0) nrm[row] = qn;
  }
}

// ---------------------------------------------------------------------------
// TDM: DMA one contiguous 16 KB K-tile (128 rows x 64 halves) into LDS.
// D# per CDNA5 ISA ch.8: group0 = {count=1, lds_addr, 57b global addr, type=2},
// group1 = {data_size=8B, tensor/tile dims in 8B units}. Groups 2/3 zero (2D).
// This toolchain exposes the 6-arg builtin (g0, g1, g2, g3, extra, cpol).
// ---------------------------------------------------------------------------
#if __has_builtin(__builtin_amdgcn_tensor_load_to_lds)
#define CA_HAVE_TDM 1
static __device__ __forceinline__ void tdm_load_16kb(unsigned lds_off,
                                                     unsigned long long gaddr) {
  u32x4 g0;
  g0[0] = 1u;                                              // count=1 (valid), user mode
  g0[1] = lds_off;                                         // lds_addr (bytes)
  g0[2] = (unsigned)gaddr;                                 // global_addr[31:0]
  g0[3] = (unsigned)((gaddr >> 32) & 0x01FFFFFFu) | 0x80000000u;  // addr[56:32] | type=2

  // 2048 x 8-byte units, single row; stride = 2048 units.
  i32x8 g1;
  g1[0] = 3 << 16;            // workgroup_mask=0, data_size=3 (8B), no flags
  g1[1] = (2048 & 0xFFFF) << 16;   // tensor_dim0[15:0] in bits 63:48
  g1[2] = (2048 >> 16) | (1 << 16);// tensor_dim0[31:16] | tensor_dim1[15:0]
  g1[3] = (0) | (2048 << 16);      // tensor_dim1[31:16] | tile_dim0
  g1[4] = 1;                       // tile_dim1=1, tile_dim2=0
  g1[5] = 2048;                    // tensor_dim0_stride[31:0]
  g1[6] = 0;                       // stride0[47:32] | stride1[15:0]
  g1[7] = 0;                       // tensor_dim1_stride[47:16]

  i32x4 z4 = {0, 0, 0, 0};
  i32x8 z8 = {0, 0, 0, 0, 0, 0, 0, 0};
  __builtin_amdgcn_tensor_load_to_lds(g0, g1, z4, z4, z8, 0);
}
#else
#define CA_HAVE_TDM 0
#endif

static __device__ __forceinline__ void wait_tensorcnt0() {
#if __has_builtin(__builtin_amdgcn_s_wait_tensorcnt)
  __builtin_amdgcn_s_wait_tensorcnt(0);
#else
  asm volatile("s_wait_tensorcnt 0" ::: "memory");
#endif
}

// ---------------------------------------------------------------------------
// Kernel 2: flash-style Gaussian-kernel attention.
// Grid: 16 heads x 16 query-tiles; block = 8 waves; wave owns 16 queries.
// K tile is DMA'd to LDS once per block by the TDM (vs 8x redundant vmem);
// logit = (2 q.k - |k|^2)/tau, online softmax, O += P @ V, epilogue restores
// the exact reference normalization (incl. +1e-9) using stored |q|^2.
// ---------------------------------------------------------------------------
__global__ __launch_bounds__(256, 1) void _ca_attn_kernel(
    const half_t* __restrict__ qh, const half_t* __restrict__ kh,
    const half_t* __restrict__ vt,                     // (b, d, s) f16 == V^T
    const float* __restrict__ qn_arr, const float* __restrict__ kn_arr,
    const float* __restrict__ rtau, float* __restrict__ out) {
  __shared__ half_t Kt[128 * QK];                      // 16 KB staged K tile
  __shared__ half_t Pbuf[8 * 16 * PSTR];               // wave-private P tiles (~34 KB)
  int w = threadIdx.x >> 5, l = threadIdx.x & 31;
  int lo = l & 15, hi = l >> 4;
  int b  = blockIdx.x >> 4;
  int qt = blockIdx.x & 15;
  int s0 = qt * 128 + w * 16;

  float tau  = log1pf(__expf(rtau[0])) + 1e-9f;        // softplus
  float itau = 1.0f / tau;
  float t2   = 2.0f * itau;

  const half_t* qrow = qh + (size_t)(b * SEQ + s0 + lo) * QK;
  v16h qa0 = load_a_h(qrow, 0, hi);
  v16h qa1 = load_a_h(qrow, 32, hi);

  v8f o0 = {}, o1 = {}, o2 = {}, o3 = {};
  float m8[8], l8[8], sc[8];
#pragma unroll
  for (int r = 0; r < 8; ++r) { m8[r] = -1e30f; l8[r] = 0.0f; }

  half_t* P = Pbuf + w * (16 * PSTR);
  const half_t* vb = vt + (size_t)(b * DV) * SEQ;

#if CA_HAVE_TDM
  unsigned kt_lds_off =
      (unsigned)(unsigned long long)(__attribute__((address_space(3))) half_t*)Kt;
#endif

  for (int kt = 0; kt < 16; ++kt) {
    int j0 = kt * 128;
    // ---- stage K tile (contiguous 16 KB) into LDS ----
#if CA_HAVE_TDM
    if (w == 0) {
      tdm_load_16kb(kt_lds_off,
                    (unsigned long long)(const void*)(kh + (size_t)(b * SEQ + j0) * QK));
      wait_tensorcnt0();
    }
#else
    {
      const unsigned long long* gk = (const unsigned long long*)(kh + (size_t)(b * SEQ + j0) * QK);
      unsigned long long* lk = (unsigned long long*)Kt;
      for (int i = threadIdx.x; i < 2048; i += 256) lk[i] = gk[i];
    }
#endif
    __syncthreads();

    float sv[8][8];
    // S = Q @ K^T (2/tau scaled) minus |k|^2/tau; K B-frags come from LDS.
#pragma unroll
    for (int t = 0; t < 8; ++t) {
      const half_t* krow = Kt + (size_t)(t * 16 + lo) * QK;
      v8f c = {};
      c = WMMA_F16(qa0, load_b_h(krow, 0, hi), c);
      c = WMMA_F16(qa1, load_b_h(krow, 32, hi), c);
      float kadj = -kn_arr[b * SEQ + j0 + t * 16 + lo] * itau;
#pragma unroll
      for (int r = 0; r < 8; ++r) sv[t][r] = c[r] * t2 + kadj;
    }
    // Online softmax: row max, rescale, exp, row sum, stash P to LDS as f16.
#pragma unroll
    for (int r = 0; r < 8; ++r) {
      float mt = sv[0][r];
#pragma unroll
      for (int t = 1; t < 8; ++t) mt = fmaxf(mt, sv[t][r]);
      mt = redmax16(mt);
      float mn = fmaxf(m8[r], mt);
      sc[r] = __expf(m8[r] - mn);
      m8[r] = mn;
      float rs = 0.0f;
#pragma unroll
      for (int t = 0; t < 8; ++t) {
        float p = __expf(sv[t][r] - mn);
        rs += p;
        P[(r + hi * 8) * PSTR + t * 16 + lo] = (half_t)p;
      }
      rs = redadd16(rs);
      l8[r] = l8[r] * sc[r] + rs;
      o0[r] *= sc[r]; o1[r] *= sc[r]; o2[r] *= sc[r]; o3[r] *= sc[r];
    }
    // Cross-lane LDS dependency: wait for the wave's own ds_stores.
    asm volatile("s_wait_dscnt 0" ::: "memory");
    // O += P @ V  (V^T rows are contiguous along j -> perfect B-frags)
#pragma unroll
    for (int jc = 0; jc < 4; ++jc) {
      int c = jc * 32;
      v16h pa = load_a_h(P + lo * PSTR, c, hi);
      o0 = WMMA_F16(pa, load_b_h(vb + (size_t)(0  + lo) * SEQ + j0, c, hi), o0);
      o1 = WMMA_F16(pa, load_b_h(vb + (size_t)(16 + lo) * SEQ + j0, c, hi), o1);
      o2 = WMMA_F16(pa, load_b_h(vb + (size_t)(32 + lo) * SEQ + j0, c, hi), o2);
      o3 = WMMA_F16(pa, load_b_h(vb + (size_t)(48 + lo) * SEQ + j0, c, hi), o3);
    }
    __syncthreads();  // all waves done with Kt before next DMA overwrites it
  }
  // Epilogue: alpha_norm = p / (L + 1e-9 * exp(|q|^2/tau - m)); out is (b, d, s).
#pragma unroll
  for (int r = 0; r < 8; ++r) {
    float qn = qn_arr[b * SEQ + s0 + r + hi * 8];
    float denom = l8[r] + 1e-9f * __expf(fminf(qn * itau - m8[r], 80.0f));
    float rcp = 1.0f / denom;
    int sidx = s0 + r + hi * 8;
    float* ob = out + (size_t)(b * DV) * SEQ + sidx;
    ob[(size_t)(0  + lo) * SEQ] = o0[r] * rcp;
    ob[(size_t)(16 + lo) * SEQ] = o1[r] * rcp;
    ob[(size_t)(32 + lo) * SEQ] = o2[r] * rcp;
    ob[(size_t)(48 + lo) * SEQ] = o3[r] * rcp;
  }
}

// ---------------------------------------------------------------------------
extern "C" void kernel_launch(void* const* d_in, const int* in_sizes, int n_in,
                              void* d_out, int out_size, void* d_ws, size_t ws_size,
                              hipStream_t stream) {
  const float* q_all = (const float*)d_in[0];
  const float* k_all = (const float*)d_in[1];
  const float* xh    = (const float*)d_in[2];
  const float* Wq    = (const float*)d_in[3];
  const float* bq    = (const float*)d_in[4];
  const float* Wk    = (const float*)d_in[5];
  const float* bk    = (const float*)d_in[6];
  const float* ln_g  = (const float*)d_in[7];
  const float* ln_b  = (const float*)d_in[8];
  const float* rtau  = (const float*)d_in[9];
  float* out = (float*)d_out;

  char* ws = (char*)d_ws;
  // Workspace layout (~12.5 MB total)
  half_t* qh  = (half_t*)(ws);                                   // 4 MB
  half_t* kh  = (half_t*)(ws + (4u << 20));                      // 4 MB
  half_t* vt  = (half_t*)(ws + (8u << 20));                      // 4 MB
  half_t* wqt = (half_t*)(ws + (12u << 20));                     // 64 KB
  half_t* wkt = (half_t*)(ws + (12u << 20) + (1u << 17));        // 64 KB
  float*  qn  = (float*)(ws + (12u << 20) + (2u << 17));         // 128 KB
  float*  kn  = (float*)(ws + (12u << 20) + (3u << 17));         // 128 KB

  int nv = BH * DV * SEQ;  // 2,097,152
  _ca_prep_kernel<<<(nv + 255) / 256, 256, 0, stream>>>(Wq, Wk, xh, wqt, wkt, vt, nv);
  _ca_proj_ln_kernel<<<(BH * SEQ) / 128, 256, 0, stream>>>(q_all, wqt, bq, ln_g, ln_b, qh, qn);
  _ca_proj_ln_kernel<<<(BH * SEQ) / 128, 256, 0, stream>>>(k_all, wkt, bk, ln_g, ln_b, kh, kn);
  _ca_attn_kernel<<<BH * 16, 256, 0, stream>>>(qh, kh, vt, qn, kn, rtau, out);

  (void)in_sizes; (void)n_in; (void)out_size; (void)ws_size;
}
